// Wasserstein_Sinkhorn_14594298872054
// MI455X (gfx1250) — matrix operand order repaired and checked by reading, not compiled, not run on hardware
//
#include <hip/hip_runtime.h>

typedef __attribute__((ext_vector_type(16))) __bf16 v16bf;
typedef __attribute__((ext_vector_type(8)))  __bf16 v8bf;
typedef __attribute__((ext_vector_type(8)))  float  v8f;

#define GRID32   32
#define ELEMS    1024      // 32*32
#define NBATCH   48        // N*C
#define NTHREADS 128       // 4 wave32s
#define NITER    1000
#define CLAMPMIN 3e-4f
#define INVLAM   10.0f     // 1/0.1

// ---------- f32 operand loaders (init / final only) ----------
__device__ __forceinline__ v16bf loadA_f32(const float* L, int tm, int l15, int h) {
    v16bf a;
#pragma unroll
    for (int e = 0; e < 16; ++e) {
        const int ka = (e < 8) ? (8 * h + e) : (8 + 8 * h + e);
        a[e] = (__bf16)L[(16 * tm + l15) * GRID32 + ka];
    }
    return a;
}
__device__ __forceinline__ v16bf loadB_f32(const float* R, int tn, int l15, int h) {
    v16bf b;
#pragma unroll
    for (int e = 0; e < 16; ++e)
        b[e] = (__bf16)R[((e < 8) ? (8 * h + e) : (8 + 8 * h + e)) * GRID32 + 16 * tn + l15];
    return b;
}

// ---------- bf16 operand loaders (hot loop; wide, convert-free) ----------
// A-operand of X (array stored [i*32+j] = X[i][j]): row m=16*tm+l15, two
// contiguous 16B K-chunks at +8h and +16+8h.
__device__ __forceinline__ v16bf loadA_bf16(const __bf16* X, int tm, int l15, int h) {
    const __bf16* base = X + (16 * tm + l15) * GRID32 + 8 * h;
    v8bf lo = *(const v8bf*)(base);
    v8bf hi = *(const v8bf*)(base + 16);
    return __builtin_shufflevector(lo, hi, 0,1,2,3,4,5,6,7,8,9,10,11,12,13,14,15);
}
// B-operand of V from col-major storage (array [n*32+k] = V[k][n]):
// col n=16*tn+l15, two contiguous 16B K-chunks at +8h and +16+8h.
__device__ __forceinline__ v16bf loadB_bf16(const __bf16* X, int tn, int l15, int h) {
    const __bf16* base = X + (16 * tn + l15) * GRID32 + 8 * h;
    v8bf lo = *(const v8bf*)(base);
    v8bf hi = *(const v8bf*)(base + 16);
    return __builtin_shufflevector(lo, hi, 0,1,2,3,4,5,6,7,8,9,10,11,12,13,14,15);
}

__device__ __forceinline__ v8f wmma_bf16(v16bf a, v16bf b) {
    v8f c = {0.f, 0.f, 0.f, 0.f, 0.f, 0.f, 0.f, 0.f};
    return __builtin_amdgcn_wmma_f32_16x16x32_bf16(false, a, false, b,
                                                   (short)0, c, false, false);
}

// Transposed packed store of a C/D tile: element (row=16tm+8h+r, col=16tn+l15)
// of P goes to X[col*32+row] (8 consecutive bf16 -> one 16B store).
// If P = M^T this makes X row-major M; if P = M it makes X col-major M.
__device__ __forceinline__ void storeT_bf16(__bf16* X, v8f c, int tm, int tn, int l15, int h) {
    v8bf d;
#pragma unroll
    for (int r = 0; r < 8; ++r) d[r] = (__bf16)c[r];
    *(v8bf*)(X + (16 * tn + l15) * GRID32 + 16 * tm + 8 * h) = d;
}
// Row-major scattered store (final contraction only).
__device__ __forceinline__ void storeRM_bf16(__bf16* X, v8f c, int tm, int tn, int l15, int h) {
#pragma unroll
    for (int r = 0; r < 8; ++r)
        X[(16 * tm + 8 * h + r) * GRID32 + 16 * tn + l15] = (__bf16)c[r];
}

__global__ void __launch_bounds__(NTHREADS)
sinkhorn_wmma_kernel(const float* __restrict__ in0,
                     const float* __restrict__ in1,
                     float* __restrict__ ws) {
    const int b    = blockIdx.x;            // batch row (n*C + c)
    const int tid  = threadIdx.x;
    const int lane = tid & 31;
    const int wave = tid >> 5;
    const int tm   = (wave >> 1) & 1;
    const int tn   = wave & 1;
    const int l15  = lane & 15;
    const int h    = lane >> 4;

    __shared__ float Amat[ELEMS];                    // exp(-|p-q|/lam)
    __shared__ float Gmat[ELEMS];                    // |p-q| * Amat
    __shared__ __align__(16) __bf16 Ub0[ELEMS], Ub1[ELEMS];  // col-major U (ping-pong)
    __shared__ __align__(16) __bf16 Vb0[ELEMS], Vb1[ELEMS];  // col-major V (ping-pong)
    __shared__ __align__(16) __bf16 Tb[ELEMS];               // row-major T = A*X
    __shared__ float red[NTHREADS];
    __shared__ int   badf[2];

    // ---- build separable kernel factors + initial u, v (bf16 1/1024 exact) ----
    const __bf16 uinit = (__bf16)(1.0f / (float)ELEMS);
    for (int i = tid; i < ELEMS; i += NTHREADS) {
        const int p = i >> 5, q = i & 31;
        const float d  = fabsf((float)(p - q));
        const float av = expf(-INVLAM * d);
        Amat[i] = av;
        Gmat[i] = av * d;
        Ub0[i] = uinit;
        Vb0[i] = uinit;
    }
    if (tid == 0) { badf[0] = 0; badf[1] = 0; }

    // ---- load a, b directly in C/D register layout; clamp + row-sum ----
    v8f ar, br;
    float pa = 0.f, pb = 0.f;
#pragma unroll
    for (int r = 0; r < 8; ++r) {
        const int i = (16 * tm + 8 * h + r) * GRID32 + 16 * tn + l15;
        float x0 = fmaxf(in0[b * ELEMS + i] * 0.5f + 0.5f, CLAMPMIN);
        float x1 = fmaxf(in1[b * ELEMS + i] * 0.5f + 0.5f, CLAMPMIN);
        ar[r] = x0; br[r] = x1;
        pa += x0;  pb += x1;
    }
    red[tid] = pa; __syncthreads();
    for (int s = NTHREADS / 2; s > 0; s >>= 1) {
        if (tid < s) red[tid] += red[tid + s];
        __syncthreads();
    }
    const float rsa = 1.f / (red[0] + 1e-35f);
    __syncthreads();
    red[tid] = pb; __syncthreads();
    for (int s = NTHREADS / 2; s > 0; s >>= 1) {
        if (tid < s) red[tid] += red[tid + s];
        __syncthreads();
    }
    const float rsb = 1.f / (red[0] + 1e-35f);
    __syncthreads();
#pragma unroll
    for (int r = 0; r < 8; ++r) { ar[r] *= rsa; br[r] *= rsb; }

    // Single loop-invariant operand: every loop GEMM is (X * A), B-op = Amat.
    const v16bf amatB = loadB_f32(Amat, tn, l15, h);

    const __bf16* Ucb = Ub0; __bf16* Unb = Ub1;
    const __bf16* Vcb = Vb0; __bf16* Vnb = Vb1;
    v8f u_keep;
#pragma unroll
    for (int r = 0; r < 8; ++r) u_keep[r] = 1.0f / (float)ELEMS;

    // ---- Sinkhorn: uK = A*U*A, computed as (U^T*A) then (T*A); A symmetric ----
    for (int it = 0; it < NITER; ++it) {
        const int p = it & 1;

        // Tb(row-major T=A*U) <- transpose-store of (Uc^T * A)
        storeT_bf16(Tb, wmma_bf16(loadA_bf16(Ucb, tm, l15, h), amatB), tm, tn, l15, h);
        __syncthreads();

        // w = T*A = uK ;  v_new = b * rcp(w)
        v8f w = wmma_bf16(loadA_bf16(Tb, tm, l15, h), amatB);
        v8f vn;
        int mybad = 0;
#pragma unroll
        for (int r = 0; r < 8; ++r) {
            vn[r] = br[r] * __builtin_amdgcn_rcpf(w[r]);
            if (!(fabsf(vn[r]) < __builtin_inff())) mybad = 1;
        }
        storeT_bf16(Vnb, vn, tm, tn, l15, h);   // col-major v_new
        __syncthreads();

        // Tb(row-major T=A*Vn) <- transpose-store of (Vn^T * A)
        storeT_bf16(Tb, wmma_bf16(loadA_bf16(Vnb, tm, l15, h), amatB), tm, tn, l15, h);
        __syncthreads();

        // w = T*A = v_new K^T ;  u_new = a * rcp(w)
        w = wmma_bf16(loadA_bf16(Tb, tm, l15, h), amatB);
        v8f un;
#pragma unroll
        for (int r = 0; r < 8; ++r) {
            un[r] = ar[r] * __builtin_amdgcn_rcpf(w[r]);
            if (!(fabsf(un[r]) < __builtin_inff())) mybad = 1;
        }
        storeT_bf16(Unb, un, tm, tn, l15, h);   // col-major u_new
        if (mybad) atomicOr(&badf[p], 1);
        __syncthreads();

        // commit via ping-pong (uniform); on bad, keep old u AND v
        const int isbad = badf[p];
        if (tid == 0) badf[p ^ 1] = 0;
        if (!isbad) {
            const __bf16* t;
            t = Ucb; Ucb = Unb; Unb = (__bf16*)t;
            t = Vcb; Vcb = Vnb; Vnb = (__bf16*)t;
            u_keep = un;
        }
    }

    // ---- dist = <U, G*V*A> + <U, A*V*G>  (K.M = G(x)A + A(x)G) ----
    const v16bf vB = loadB_bf16(Vcb, tn, l15, h);  // col-major V == B-operand order
    float part = 0.f;

    storeRM_bf16(Tb, wmma_bf16(loadA_f32(Gmat, tm, l15, h), vB), tm, tn, l15, h);
    __syncthreads();
    {
        v8f w = wmma_bf16(loadA_bf16(Tb, tm, l15, h), amatB);
#pragma unroll
        for (int r = 0; r < 8; ++r) part += u_keep[r] * w[r];
    }
    __syncthreads();   // all waves done with Tb before overwrite

    storeRM_bf16(Tb, wmma_bf16(loadA_f32(Amat, tm, l15, h), vB), tm, tn, l15, h);
    __syncthreads();
    {
        v8f w = wmma_bf16(loadA_bf16(Tb, tm, l15, h), loadB_f32(Gmat, tn, l15, h));
#pragma unroll
        for (int r = 0; r < 8; ++r) part += u_keep[r] * w[r];
    }
    __syncthreads();

    red[tid] = part; __syncthreads();
    for (int s = NTHREADS / 2; s > 0; s >>= 1) {
        if (tid < s) red[tid] += red[tid + s];
        __syncthreads();
    }
    if (tid == 0) ws[b] = red[0];
}

// Deterministic channel reduction: out[n] = ws[3n] + ws[3n+1] + ws[3n+2]
__global__ void sinkhorn_finalize_kernel(const float* __restrict__ ws,
                                         float* __restrict__ out) {
    const int n = threadIdx.x;
    if (n < 16)
        out[n] = ws[3 * n] + ws[3 * n + 1] + ws[3 * n + 2];
}

extern "C" void kernel_launch(void* const* d_in, const int* in_sizes, int n_in,
                              void* d_out, int out_size, void* d_ws, size_t ws_size,
                              hipStream_t stream) {
    const float* in0 = (const float*)d_in[0];
    const float* in1 = (const float*)d_in[1];
    // d_in[2] (M) is not needed: it is reconstructed analytically via separability.
    float* ws  = (float*)d_ws;     // 48 floats of scratch
    float* out = (float*)d_out;    // 16 floats

    sinkhorn_wmma_kernel<<<NBATCH, NTHREADS, 0, stream>>>(in0, in1, ws);
    sinkhorn_finalize_kernel<<<1, 16, 0, stream>>>(ws, out);
}
